// LoRANode2Edge2NodeBlock_83734682402871
// MI455X (gfx1250) — matrix operand order repaired and checked by reading, not compiled, not run on hardware
//
#include <hip/hip_runtime.h>
#include <hip/hip_bf16.h>

#define D 128

typedef __attribute__((ext_vector_type(16))) __bf16 v16bf;
typedef __attribute__((ext_vector_type(8)))  float  v8f;

// fp32 -> bf16 RNE bit trick (used only in memory-bound prep kernels)
__device__ __forceinline__ unsigned short f2bf(float f) {
  union { float f; unsigned u; } x; x.f = f;
  unsigned u = x.u;
  u += 0x7FFFu + ((u >> 16) & 1u);
  return (unsigned short)(u >> 16);
}

__device__ __forceinline__ v8f wmma_bf16(v16bf a, v16bf b, v8f c) {
  return __builtin_amdgcn_wmma_f32_16x16x32_bf16(
      false, a, false, b, (short)0, c, false, false);
}

// A-fragment (16x32 bf16) from a bf16 row: lane holds K = {kb..kb+7, kb+16..kb+23},
// i.e. two 8-element consecutive runs -> two b128 loads, no conversion.
__device__ __forceinline__ v16bf load_a_bf(const unsigned short* __restrict__ row, int kt, int half) {
  union { uint4 q[2]; v16bf v; } u;
  const unsigned short* p = row + kt * 32 + half * 8;
  u.q[0] = *(const uint4*)(p);
  u.q[1] = *(const uint4*)(p + 16);
  return u.v;
}

// A-fragment from an fp32 row (edge / agg): two 8-float runs + hw bf16 casts.
__device__ __forceinline__ v16bf load_a_f32(const float* __restrict__ row, int kt, int half) {
  union { v16bf v; __bf16 b[16]; } u;
  const float* p = row + kt * 32 + half * 8;
  float4 f0 = *(const float4*)(p);
  float4 f1 = *(const float4*)(p + 4);
  float4 f2 = *(const float4*)(p + 16);
  float4 f3 = *(const float4*)(p + 20);
  u.b[0]  = (__bf16)f0.x; u.b[1]  = (__bf16)f0.y; u.b[2]  = (__bf16)f0.z; u.b[3]  = (__bf16)f0.w;
  u.b[4]  = (__bf16)f1.x; u.b[5]  = (__bf16)f1.y; u.b[6]  = (__bf16)f1.z; u.b[7]  = (__bf16)f1.w;
  u.b[8]  = (__bf16)f2.x; u.b[9]  = (__bf16)f2.y; u.b[10] = (__bf16)f2.z; u.b[11] = (__bf16)f2.w;
  u.b[12] = (__bf16)f3.x; u.b[13] = (__bf16)f3.y; u.b[14] = (__bf16)f3.z; u.b[15] = (__bf16)f3.w;
  return u.v;
}

// Pre-packed B-fragment: each lane's 16 bf16 contiguous -> 2x ds_load_b128.
__device__ __forceinline__ v16bf load_b_frag(const unsigned short* w, int kt, int nt, int lane) {
  const uint4* p = (const uint4*)(w + ((((kt * 8) + nt) * 32 + lane) << 4));
  union { uint4 q[2]; v16bf v; } u;
  u.q[0] = p[0];
  u.q[1] = p[1];
  return u.v;
}

__device__ __forceinline__ float silu_f(float x) { return x / (1.0f + __expf(-x)); }

// Async global->LDS 16B copy (CDNA5 GLOBAL_LOAD_ASYNC_TO_LDS_B128, ASYNCcnt-tracked).
// lds_off = low 32 bits of a generic LDS pointer (== LDS byte address).
__device__ __forceinline__ void async_g2l_b128(unsigned lds_off, const void* g) {
  asm volatile("global_load_async_to_lds_b128 %0, %1, off"
               :: "v"(lds_off), "v"((unsigned long long)g) : "memory");
}
__device__ __forceinline__ void wait_async0() {
  asm volatile("s_wait_asynccnt 0" ::: "memory");
}

// Pack 5 weights [D,D] fp32 row-major -> bf16 WMMA-B layout. B[k][n] = W[n][k].
__global__ void prep_weights_kernel(const float* __restrict__ W0, const float* __restrict__ W1,
                                    const float* __restrict__ W2, const float* __restrict__ W3,
                                    const float* __restrict__ W4, unsigned short* __restrict__ packed) {
  int idx = blockIdx.x * blockDim.x + threadIdx.x;
  if (idx >= 5 * D * D) return;
  int mat = idx >> 14;
  int r   = idx & 16383;
  int kt  = r >> 12;
  int r2  = r & 4095;
  int nt  = r2 >> 9;
  int r3  = r2 & 511;
  int lane = r3 >> 4;
  int e    = r3 & 15;
  int k = kt * 32 + (lane >> 4) * 8 + (e & 7) + (e >> 3) * 16;
  int n = nt * 16 + (lane & 15);
  const float* W = (mat == 0) ? W0 : (mat == 1) ? W1 : (mat == 2) ? W2 : (mat == 3) ? W3 : W4;
  packed[idx] = f2bf(W[n * D + k]);
}

// Row-major fp32 -> bf16 (8 elements / thread)
__global__ void cvt_bf16_kernel(const float* __restrict__ a, unsigned short* __restrict__ o, int n8) {
  int i = blockIdx.x * blockDim.x + threadIdx.x;
  if (i >= n8) return;
  const float4* p = (const float4*)a + 2 * (size_t)i;
  float4 f0 = p[0], f1 = p[1];
  union { uint4 q; __bf16 b[8]; } u;
  u.b[0] = (__bf16)f0.x; u.b[1] = (__bf16)f0.y; u.b[2] = (__bf16)f0.z; u.b[3] = (__bf16)f0.w;
  u.b[4] = (__bf16)f1.x; u.b[5] = (__bf16)f1.y; u.b[6] = (__bf16)f1.z; u.b[7] = (__bf16)f1.w;
  ((uint4*)o)[i] = u.q;
}

__global__ void zero_kernel(float4* __restrict__ p, int n4) {
  int i = blockIdx.x * blockDim.x + threadIdx.x;
  if (i < n4) p[i] = make_float4(0.f, 0.f, 0.f, 0.f);
}

// ---- STE2E: 3 fused GEMMs -> silu -> LN -> residual -> store + atomic segment-sum
__global__ void __launch_bounds__(256) edge_kernel(
    const unsigned short* __restrict__ src_bf, const unsigned short* __restrict__ tgt_bf,
    const float* __restrict__ edge,
    const int* __restrict__ src_idx, const int* __restrict__ tgt_idx,
    const unsigned short* __restrict__ packW,  // 3 matrices: s2e, t2e, e2e
    const float* __restrict__ ln_g, const float* __restrict__ ln_b,
    float* __restrict__ edge_out, float* __restrict__ agg, int E) {
  extern __shared__ char smem[];
  unsigned short* wlds = (unsigned short*)smem;                      // 3 * 32KB bf16 weights
  float* scr = (float*)(smem + 3 * D * D * sizeof(unsigned short));  // 8 waves * 16*128 f32

  {  // async weight staging: 6144 x 16B chunks, 24 full iterations for 256 threads
    const char* g = (const char*)packW;
    const unsigned lds_base = (unsigned)(size_t)smem;
    const int n16 = 3 * D * D * 2 / 16;
    for (int i = threadIdx.x; i < n16; i += blockDim.x)
      async_g2l_b128(lds_base + i * 16, g + (size_t)i * 16);
    wait_async0();
  }
  __syncthreads();

  const int wave = threadIdx.x >> 5;
  const int lane = threadIdx.x & 31;
  const int rowM = lane & 15;
  const int half = lane >> 4;
  float* myscr = scr + wave * 16 * D;

  const int tiles = (E + 15) >> 4;
  for (int tile = blockIdx.x * 8 + wave; tile < tiles; tile += gridDim.x * 8) {
    const int e  = tile * 16 + rowM;
    const bool rv = (e < E);
    const int ec = rv ? e : (E - 1);
    const int si = src_idx[ec];
    const int ti = tgt_idx[ec];
    const unsigned short* srow = src_bf + (size_t)si * D;
    const unsigned short* trow = tgt_bf + (size_t)ti * D;
    const float* erow = edge + (size_t)ec * D;

    const v8f vzero = {0.f, 0.f, 0.f, 0.f, 0.f, 0.f, 0.f, 0.f};
    v8f acc[8];
#pragma unroll
    for (int t = 0; t < 8; ++t) acc[t] = vzero;

#pragma unroll
    for (int kt = 0; kt < 4; ++kt) {
      v16bf aS = load_a_bf(srow, kt, half);
      v16bf aT = load_a_bf(trow, kt, half);
      v16bf aE = load_a_f32(erow, kt, half);
#pragma unroll
      for (int nt = 0; nt < 8; ++nt) {
        acc[nt] = wmma_bf16(aS, load_b_frag(wlds + 0 * D * D, kt, nt, lane), acc[nt]);
        acc[nt] = wmma_bf16(aT, load_b_frag(wlds + 1 * D * D, kt, nt, lane), acc[nt]);
        acc[nt] = wmma_bf16(aE, load_b_frag(wlds + 2 * D * D, kt, nt, lane), acc[nt]);
      }
    }

    // C layout: VGPR r, lanes 0-15 -> M=r, lanes 16-31 -> M=8+r; N(tile) = lane&15.
#pragma unroll
    for (int nt = 0; nt < 8; ++nt) {
#pragma unroll
      for (int r = 0; r < 8; ++r) {
        myscr[(half * 8 + r) * D + nt * 16 + rowM] = silu_f(acc[nt][r]);
      }
    }
    asm volatile("s_wait_dscnt 0" ::: "memory");  // same-wave LDS RAW

    // each half-lane reduces its 64 cols; combine halves with one shuffle
    const int cbase = half * 64;
    float s0 = 0.f, s1 = 0.f;
    float4 hv[16];
#pragma unroll
    for (int i = 0; i < 16; ++i) {
      float4 h = *(const float4*)(myscr + rowM * D + cbase + 4 * i);
      hv[i] = h;
      s0 += h.x + h.y + h.z + h.w;
      s1 += h.x * h.x + h.y * h.y + h.z * h.z + h.w * h.w;
    }
    s0 += __shfl_xor(s0, 16, 32);
    s1 += __shfl_xor(s1, 16, 32);
    const float mean = s0 * (1.0f / D);
    const float var  = s1 * (1.0f / D) - mean * mean;
    const float rstd = rsqrtf(var + 1e-5f);

    if (rv) {
      float* orow = edge_out + (size_t)e * D;
      float* arow = agg + (size_t)ti * D;
#pragma unroll
      for (int i = 0; i < 16; ++i) {
        const int c = cbase + i * 4;
        float4 h  = hv[i];
        float4 gg = *(const float4*)(ln_g + c);
        float4 bb = *(const float4*)(ln_b + c);
        float4 ei = *(const float4*)(erow + c);
        float4 o;
        o.x = ei.x + (h.x - mean) * rstd * gg.x + bb.x;
        o.y = ei.y + (h.y - mean) * rstd * gg.y + bb.y;
        o.z = ei.z + (h.z - mean) * rstd * gg.z + bb.z;
        o.w = ei.w + (h.w - mean) * rstd * gg.w + bb.w;
        *(float4*)(orow + c) = o;
        atomicAdd(arow + c + 0, o.x);
        atomicAdd(arow + c + 1, o.y);
        atomicAdd(arow + c + 2, o.z);
        atomicAdd(arow + c + 3, o.w);
      }
    }
  }
}

// ---- ET2T: 2 fused GEMMs -> silu -> LN -> residual
__global__ void __launch_bounds__(256) node_kernel(
    const float* __restrict__ tgt, const unsigned short* __restrict__ tgt_bf,
    const float* __restrict__ agg,
    const unsigned short* __restrict__ packW,  // 2 matrices: e2t, t2t
    const float* __restrict__ ln_g, const float* __restrict__ ln_b,
    float* __restrict__ tgt_out, int N) {
  extern __shared__ char smem[];
  unsigned short* wlds = (unsigned short*)smem;
  float* scr = (float*)(smem + 2 * D * D * sizeof(unsigned short));

  {
    const char* g = (const char*)packW;
    const unsigned lds_base = (unsigned)(size_t)smem;
    const int n16 = 2 * D * D * 2 / 16;
    for (int i = threadIdx.x; i < n16; i += blockDim.x)
      async_g2l_b128(lds_base + i * 16, g + (size_t)i * 16);
    wait_async0();
  }
  __syncthreads();

  const int wave = threadIdx.x >> 5;
  const int lane = threadIdx.x & 31;
  const int rowM = lane & 15;
  const int half = lane >> 4;
  float* myscr = scr + wave * 16 * D;

  const int tiles = (N + 15) >> 4;
  for (int tile = blockIdx.x * 8 + wave; tile < tiles; tile += gridDim.x * 8) {
    const int n  = tile * 16 + rowM;
    const bool rv = (n < N);
    const int nc = rv ? n : (N - 1);
    const float* arow = agg + (size_t)nc * D;
    const unsigned short* trow = tgt_bf + (size_t)nc * D;
    const float* trow32 = tgt + (size_t)nc * D;

    const v8f vzero = {0.f, 0.f, 0.f, 0.f, 0.f, 0.f, 0.f, 0.f};
    v8f acc[8];
#pragma unroll
    for (int t = 0; t < 8; ++t) acc[t] = vzero;

#pragma unroll
    for (int kt = 0; kt < 4; ++kt) {
      v16bf aA = load_a_f32(arow, kt, half);
      v16bf aT = load_a_bf(trow, kt, half);
#pragma unroll
      for (int nt = 0; nt < 8; ++nt) {
        acc[nt] = wmma_bf16(aA, load_b_frag(wlds + 0 * D * D, kt, nt, lane), acc[nt]);
        acc[nt] = wmma_bf16(aT, load_b_frag(wlds + 1 * D * D, kt, nt, lane), acc[nt]);
      }
    }

#pragma unroll
    for (int nt = 0; nt < 8; ++nt) {
#pragma unroll
      for (int r = 0; r < 8; ++r) {
        myscr[(half * 8 + r) * D + nt * 16 + rowM] = silu_f(acc[nt][r]);
      }
    }
    asm volatile("s_wait_dscnt 0" ::: "memory");

    const int cbase = half * 64;
    float s0 = 0.f, s1 = 0.f;
    float4 hv[16];
#pragma unroll
    for (int i = 0; i < 16; ++i) {
      float4 h = *(const float4*)(myscr + rowM * D + cbase + 4 * i);
      hv[i] = h;
      s0 += h.x + h.y + h.z + h.w;
      s1 += h.x * h.x + h.y * h.y + h.z * h.z + h.w * h.w;
    }
    s0 += __shfl_xor(s0, 16, 32);
    s1 += __shfl_xor(s1, 16, 32);
    const float mean = s0 * (1.0f / D);
    const float var  = s1 * (1.0f / D) - mean * mean;
    const float rstd = rsqrtf(var + 1e-5f);

    if (rv) {
      float* orow = tgt_out + (size_t)n * D;
#pragma unroll
      for (int i = 0; i < 16; ++i) {
        const int c = cbase + i * 4;
        float4 h  = hv[i];
        float4 gg = *(const float4*)(ln_g + c);
        float4 bb = *(const float4*)(ln_b + c);
        float4 ti = *(const float4*)(trow32 + c);
        float4 o;
        o.x = ti.x + (h.x - mean) * rstd * gg.x + bb.x;
        o.y = ti.y + (h.y - mean) * rstd * gg.y + bb.y;
        o.z = ti.z + (h.z - mean) * rstd * gg.z + bb.z;
        o.w = ti.w + (h.w - mean) * rstd * gg.w + bb.w;
        *(float4*)(orow + c) = o;
      }
    }
  }
}

extern "C" void kernel_launch(void* const* d_in, const int* in_sizes, int n_in,
                              void* d_out, int out_size, void* d_ws, size_t ws_size,
                              hipStream_t stream) {
  const float* src     = (const float*)d_in[0];
  const float* tgt     = (const float*)d_in[1];
  const float* edge    = (const float*)d_in[2];
  const int*   src_idx = (const int*)d_in[3];
  const int*   tgt_idx = (const int*)d_in[4];
  const float* W_s2e   = (const float*)d_in[5];
  const float* W_t2e   = (const float*)d_in[6];
  const float* W_e2e   = (const float*)d_in[7];
  const float* W_e2t   = (const float*)d_in[8];
  const float* W_t2t   = (const float*)d_in[9];
  const float* ln1_g   = (const float*)d_in[10];
  const float* ln1_b   = (const float*)d_in[11];
  const float* ln2_g   = (const float*)d_in[12];
  const float* ln2_b   = (const float*)d_in[13];

  const int N = in_sizes[0] / D;
  const int E = in_sizes[3];

  // ws layout: agg f32 [N,D] | packW bf16 [5*D*D] | src_bf [N,D] | tgt_bf [N,D]
  char* ws = (char*)d_ws;
  float* agg = (float*)ws;                       ws += (size_t)N * D * sizeof(float);
  unsigned short* packW = (unsigned short*)ws;   ws += (size_t)5 * D * D * sizeof(unsigned short);
  unsigned short* src_bf = (unsigned short*)ws;  ws += (size_t)N * D * sizeof(unsigned short);
  unsigned short* tgt_bf = (unsigned short*)ws;

  prep_weights_kernel<<<(5 * D * D + 255) / 256, 256, 0, stream>>>(
      W_s2e, W_t2e, W_e2e, W_e2t, W_t2t, packW);

  const int n8 = N * D / 8;
  cvt_bf16_kernel<<<(n8 + 255) / 256, 256, 0, stream>>>(src, src_bf, n8);
  cvt_bf16_kernel<<<(n8 + 255) / 256, 256, 0, stream>>>(tgt, tgt_bf, n8);

  const int n4 = N * D / 4;
  zero_kernel<<<(n4 + 255) / 256, 256, 0, stream>>>((float4*)agg, n4);

  // edge stage: LDS = 96KB weights + 64KB scratch
  const size_t ldsE = (size_t)3 * D * D * 2 + (size_t)8 * 16 * D * 4;
  edge_kernel<<<1024, 256, ldsE, stream>>>(src_bf, tgt_bf, edge, src_idx, tgt_idx, packW,
                                           ln1_g, ln1_b, (float*)d_out, agg, E);

  // node stage: LDS = 64KB weights + 64KB scratch
  const size_t ldsN = (size_t)2 * D * D * 2 + (size_t)8 * 16 * D * 4;
  node_kernel<<<512, 256, ldsN, stream>>>(tgt, tgt_bf, agg, packW + 3 * D * D, ln2_g, ln2_b,
                                          (float*)d_out + (size_t)E * D, N);
}
